// Attention_19507741458541
// MI455X (gfx1250) — matrix operand order repaired and checked by reading, not compiled, not run on hardware
//
#include <hip/hip_runtime.h>

// Attention block for MI455X (gfx1250, wave32, WMMA).
// B=2, N=2048, C=1024, H=16, Dh=64. All GEMMs via v_wmma_f32_16x16x32_bf16.
// V is stored transposed per head so every WMMA operand load is a contiguous
// 16/32-byte vector load (no strided gathers in the flash-attention loop).

#define BATCH 2
#define SEQ   2048
#define DIM   1024
#define HEADS 16
#define DH    64
#define TOKENS (BATCH * SEQ)   // 4096

typedef __attribute__((ext_vector_type(16))) __bf16 v16bf;
typedef __attribute__((ext_vector_type(8)))  __bf16 v8bf;
typedef __attribute__((ext_vector_type(8)))  float  v8f;

union U16BF { v16bf v; v8bf h[2]; __bf16 e[16]; };
union U8F   { v8f v; float e[8]; };

static __device__ __forceinline__ v8f wmma_bf16(v16bf a, v16bf b, v8f c) {
    return __builtin_amdgcn_wmma_f32_16x16x32_bf16(
        /*neg_a=*/false, a, /*neg_b=*/false, b,
        /*c_mod=*/(short)0, c, /*reuse_a=*/false, /*reuse_b=*/false);
}

// A fragment (16x32, row-major source). ISA layout: lanes 0-15 -> M=lane,
// K = {0..7, 16..23}; lanes 16-31 -> M=lane-16, K = {8..15, 24..31}.
static __device__ __forceinline__ v16bf load_a_frag(const __bf16* A, int lda,
                                                    int m0, int k0, int lane) {
    int row = m0 + (lane & 15);
    int kb  = k0 + ((lane >> 4) << 3);
    const __bf16* p = A + (size_t)row * lda + kb;
    U16BF u;
    u.h[0] = *(const v8bf*)(p);
    u.h[1] = *(const v8bf*)(p + 16);
    return u.v;
}

// B fragment (32x16) from a transposed (column-major-as-rows) matrix Bt[N][K]:
// lanes 0-15 -> N=lane, K={0..7, 16..23}; lanes 16-31 -> N=lane-16, K={8..15, 24..31}.
static __device__ __forceinline__ v16bf load_bT_frag(const __bf16* Bt, int ldbt,
                                                     int n0, int k0, int lane) {
    int col = n0 + (lane & 15);
    int kb  = k0 + ((lane >> 4) << 3);
    const __bf16* p = Bt + (size_t)col * ldbt + kb;
    U16BF u;
    u.h[0] = *(const v8bf*)(p);
    u.h[1] = *(const v8bf*)(p + 16);
    return u.v;
}

// ---------------- Kernel 1: LayerNorm -> bf16 ----------------
__global__ void ln_kernel(const float* __restrict__ x,
                          const float* __restrict__ gamma,
                          const float* __restrict__ beta,
                          __bf16* __restrict__ xn) {
    int warp = threadIdx.x >> 5, lane = threadIdx.x & 31;
    int row = blockIdx.x * 4 + warp;
    const float* xr = x + (size_t)row * DIM;
    float vals[32];
    float s = 0.f, s2 = 0.f;
#pragma unroll
    for (int i = 0; i < 32; ++i) {
        float v = xr[lane + i * 32];
        vals[i] = v; s += v; s2 += v * v;
    }
#pragma unroll
    for (int m = 16; m >= 1; m >>= 1) {
        s  += __shfl_xor(s,  m, 32);
        s2 += __shfl_xor(s2, m, 32);
    }
    float mean = s  * (1.0f / DIM);
    float var  = s2 * (1.0f / DIM) - mean * mean;   // biased var (jnp default)
    float rinv = rsqrtf(var + 1e-5f);
    __bf16* o = xn + (size_t)row * DIM;
#pragma unroll
    for (int i = 0; i < 32; ++i) {
        int idx = lane + i * 32;
        o[idx] = (__bf16)((vals[i] - mean) * rinv * gamma[idx] + beta[idx]);
    }
}

// ------- Kernel 2: transpose + convert weights [K][N]f32 -> [N][K]bf16 -------
__global__ void cvt_t_kernel(const float* __restrict__ in, __bf16* __restrict__ out,
                             int K, int N) {
    int tid = blockIdx.x * 256 + threadIdx.x;
    if (tid >= K * N) return;
    int n = tid / K;
    int k = tid - n * K;
    out[tid] = (__bf16)in[(size_t)k * N + n];
}

// ---------------- Kernel 3: QKV GEMM -> per-head Q/K/V ----------------
// xn [4096][1024] bf16, wt [3072][1024] bf16 (transposed W_qkv)
// Q,K stored [s][B][H][SEQ][DH]; V stored transposed [2][B][H][DH][SEQ].
__global__ void qkv_gemm_kernel(const __bf16* __restrict__ xn,
                                const __bf16* __restrict__ wt,
                                __bf16* __restrict__ qkv) {
    int warp = threadIdx.x >> 5, lane = threadIdx.x & 31;
    int w  = blockIdx.x * 4 + warp;
    int tm = w / 48, tn = w % 48;           // 256 x 48 wave tiles
    int m0 = tm * 16, n0 = tn * 64;
    v8f c[4];
#pragma unroll
    for (int f = 0; f < 4; ++f) { U8F z;
#pragma unroll
        for (int v = 0; v < 8; ++v) z.e[v] = 0.f; c[f] = z.v; }

    for (int k0 = 0; k0 < DIM; k0 += 32) {
        v16bf a = load_a_frag(xn, DIM, m0, k0, lane);
#pragma unroll
        for (int f = 0; f < 4; ++f) {
            v16bf b = load_bT_frag(wt, DIM, n0 + f * 16, k0, lane);
            c[f] = wmma_bf16(a, b, c[f]);
        }
    }
#pragma unroll
    for (int f = 0; f < 4; ++f) {
        U8F u; u.v = c[f];
        int j = n0 + f * 16 + (lane & 15);
        int s = j >> 10, rem = j & 1023;
        int h = rem >> 6, d = rem & 63;
#pragma unroll
        for (int v = 0; v < 8; ++v) {
            int m  = m0 + v + ((lane >> 4) << 3);
            int bi = m >> 11, n = m & 2047;
            size_t headbase = (((size_t)s * BATCH + bi) * HEADS + h) * (SEQ * DH);
            if (s == 2) {
                // V transposed: [d][n] within the head
                qkv[headbase + (size_t)d * SEQ + n] = (__bf16)u.e[v];
            } else {
                qkv[headbase + (size_t)n * DH + d] = (__bf16)u.e[v];
            }
        }
    }
}

// ---------------- Kernel 4: flash attention ----------------
// One wave = 16 query rows of one (b,h). Key tiles of 32.
__global__ void attn_kernel(const __bf16* __restrict__ qkv,
                            __bf16* __restrict__ aout) {
    __shared__ __bf16 plds[4][16 * 32];     // per-wave P staging tile
    int warp = threadIdx.x >> 5, lane = threadIdx.x & 31;
    int w  = blockIdx.x * 4 + warp;
    int qt = w & 127, hh = (w >> 7) & 15, b = w >> 11;
    int q0 = qt * 16;

    const size_t headsz = (size_t)SEQ * DH;
    const __bf16* Qh = qkv + (((size_t)0 * BATCH + b) * HEADS + hh) * headsz;
    const __bf16* Kh = qkv + (((size_t)1 * BATCH + b) * HEADS + hh) * headsz;
    const __bf16* Vt = qkv + (((size_t)2 * BATCH + b) * HEADS + hh) * headsz; // [DH][SEQ]

    v16bf qa0 = load_a_frag(Qh, DH, q0, 0,  lane);
    v16bf qa1 = load_a_frag(Qh, DH, q0, 32, lane);

    v8f o[4];
#pragma unroll
    for (int f = 0; f < 4; ++f) { U8F z;
#pragma unroll
        for (int v = 0; v < 8; ++v) z.e[v] = 0.f; o[f] = z.v; }
    float mrun[8], lrun[8];
#pragma unroll
    for (int v = 0; v < 8; ++v) { mrun[v] = -3.0e38f; lrun[v] = 0.f; }

    const float scale = 0.125f;             // Dh^-0.5
    __bf16* myp = plds[warp];
    int ncol = lane & 15;
    int moff = (lane >> 4) << 3;

    for (int kt = 0; kt < SEQ; kt += 32) {
        if (kt + 32 < SEQ) {                // prefetch next K tile (global_prefetch_b8)
            __builtin_prefetch(Kh + (size_t)(kt + 32 + (lane & 31)) * DH, 0, 0);
        }
        U8F z;
#pragma unroll
        for (int v = 0; v < 8; ++v) z.e[v] = 0.f;
        v8f s0 = z.v, s1 = z.v;
        {   // S = Q * K^T (Dh = 64 -> 2 chained WMMAs per 16x16 tile)
            v16bf kb0 = load_bT_frag(Kh, DH, kt,      0,  lane);
            v16bf kb1 = load_bT_frag(Kh, DH, kt,      32, lane);
            s0 = wmma_bf16(qa0, kb0, s0);
            s0 = wmma_bf16(qa1, kb1, s0);
            v16bf kb2 = load_bT_frag(Kh, DH, kt + 16, 0,  lane);
            v16bf kb3 = load_bT_frag(Kh, DH, kt + 16, 32, lane);
            s1 = wmma_bf16(qa0, kb2, s1);
            s1 = wmma_bf16(qa1, kb3, s1);
        }
        U8F us0, us1; us0.v = s0; us1.v = s1;
        float corr_a[8];
#pragma unroll
        for (int v = 0; v < 8; ++v) {
            float a0 = us0.e[v] * scale;
            float a1 = us1.e[v] * scale;
            float mv = fmaxf(a0, a1);       // row reduce within 16-lane half
            mv = fmaxf(mv, __shfl_xor(mv, 8, 32));
            mv = fmaxf(mv, __shfl_xor(mv, 4, 32));
            mv = fmaxf(mv, __shfl_xor(mv, 2, 32));
            mv = fmaxf(mv, __shfl_xor(mv, 1, 32));
            float mnew = fmaxf(mrun[v], mv);
            float corr = __expf(mrun[v] - mnew);
            float p0 = __expf(a0 - mnew);
            float p1 = __expf(a1 - mnew);
            float rs = p0 + p1;
            rs += __shfl_xor(rs, 8, 32);
            rs += __shfl_xor(rs, 4, 32);
            rs += __shfl_xor(rs, 2, 32);
            rs += __shfl_xor(rs, 1, 32);
            lrun[v] = lrun[v] * corr + rs;
            mrun[v] = mnew;
            corr_a[v] = corr;
            int mrow = v + moff;            // C/D layout: row = v + 8*(lane>=16)
            myp[mrow * 32 + ncol]      = (__bf16)p0;
            myp[mrow * 32 + 16 + ncol] = (__bf16)p1;
        }
#pragma unroll
        for (int f = 0; f < 4; ++f) {       // rescale running O by corr
            U8F uo; uo.v = o[f];
#pragma unroll
            for (int v = 0; v < 8; ++v) uo.e[v] *= corr_a[v];
            o[f] = uo.v;
        }
        // LDS ops are in-order within a wave (DScnt ordering) -> safe reload
        v16bf pa = load_a_frag(myp, 32, 0, 0, lane);
#pragma unroll
        for (int f = 0; f < 4; ++f) {       // O += P * V  (Vt contiguous in key)
            v16bf vb = load_bT_frag(Vt, SEQ, f * 16, kt, lane);
            o[f] = wmma_bf16(pa, vb, o[f]);
        }
    }
#pragma unroll
    for (int f = 0; f < 4; ++f) {
        U8F uo; uo.v = o[f];
        int col = hh * 64 + f * 16 + ncol;
#pragma unroll
        for (int v = 0; v < 8; ++v) {
            int n = q0 + v + moff;
            aout[((size_t)(b * SEQ + n)) * DIM + col] =
                (__bf16)(uo.e[v] / lrun[v]);
        }
    }
}

// ---------------- Kernel 5: output projection + bias -> fp32 ----------------
__global__ void out_proj_kernel(const __bf16* __restrict__ aout,
                                const __bf16* __restrict__ wt,
                                const float* __restrict__ bias,
                                float* __restrict__ out) {
    int warp = threadIdx.x >> 5, lane = threadIdx.x & 31;
    int w  = blockIdx.x * 4 + warp;
    int tm = w >> 4, tn = w & 15;           // 256 x 16 wave tiles
    int m0 = tm * 16, n0 = tn * 64;
    v8f c[4];
#pragma unroll
    for (int f = 0; f < 4; ++f) { U8F z;
#pragma unroll
        for (int v = 0; v < 8; ++v) z.e[v] = 0.f; c[f] = z.v; }

    for (int k0 = 0; k0 < DIM; k0 += 32) {
        v16bf a = load_a_frag(aout, DIM, m0, k0, lane);
#pragma unroll
        for (int f = 0; f < 4; ++f) {
            v16bf b = load_bT_frag(wt, DIM, n0 + f * 16, k0, lane);
            c[f] = wmma_bf16(a, b, c[f]);
        }
    }
#pragma unroll
    for (int f = 0; f < 4; ++f) {
        U8F u; u.v = c[f];
        int j = n0 + f * 16 + (lane & 15);
        float bj = bias[j];
#pragma unroll
        for (int v = 0; v < 8; ++v) {
            int m = m0 + v + ((lane >> 4) << 3);
            out[(size_t)m * DIM + j] = u.e[v] + bj;
        }
    }
}

extern "C" void kernel_launch(void* const* d_in, const int* in_sizes, int n_in,
                              void* d_out, int out_size, void* d_ws, size_t ws_size,
                              hipStream_t stream) {
    (void)in_sizes; (void)n_in; (void)out_size; (void)ws_size;
    const float* x     = (const float*)d_in[0];
    const float* gamma = (const float*)d_in[1];
    const float* beta  = (const float*)d_in[2];
    const float* Wqkv  = (const float*)d_in[3];
    const float* Wout  = (const float*)d_in[4];
    const float* bout  = (const float*)d_in[5];
    float* out = (float*)d_out;

    // Workspace layout (requires 48 MiB)
    char* ws = (char*)d_ws;
    __bf16* xn     = (__bf16*)(ws);                        //  8 MiB [4096][1024]
    __bf16* wqkv_t = (__bf16*)(ws + (8u  << 20));          //  6 MiB [3072][1024]
    __bf16* wout_t = (__bf16*)(ws + (14u << 20));          //  2 MiB [1024][1024]
    __bf16* qkvb   = (__bf16*)(ws + (16u << 20));          // 24 MiB [3][B][H][...]
    __bf16* attnb  = (__bf16*)(ws + (40u << 20));          //  8 MiB [4096][1024]

    ln_kernel<<<TOKENS / 4, 128, 0, stream>>>(x, gamma, beta, xn);
    cvt_t_kernel<<<(DIM * 3 * DIM + 255) / 256, 256, 0, stream>>>(Wqkv, wqkv_t, DIM, 3 * DIM);
    cvt_t_kernel<<<(DIM * DIM + 255) / 256, 256, 0, stream>>>(Wout, wout_t, DIM, DIM);
    qkv_gemm_kernel<<<(256 * 48) / 4, 128, 0, stream>>>(xn, wqkv_t, qkvb);
    attn_kernel<<<(BATCH * HEADS * (SEQ / 16)) / 4, 128, 0, stream>>>(qkvb, attnb);
    out_proj_kernel<<<(256 * 16) / 4, 128, 0, stream>>>(attnb, wout_t, bout, out);
}
